// VectorQuantizer_73315091743020
// MI455X (gfx1250) — compile-verified, matrix operand order checked by reading, not run on hardware
//
#include <hip/hip_runtime.h>
#include <hip/hip_bf16.h>
#include <math.h>

typedef __attribute__((ext_vector_type(16))) _Float16 v16h;
typedef __attribute__((ext_vector_type(8)))  _Float16 v8h;
typedef __attribute__((ext_vector_type(8)))  float    v8f;

#define N_ROWS   32768
#define EMB_D    256
#define KCODES   1024
#define BETA_F   0.25f
#define CB_SCALE 1024.0f

// workspace layout (bytes)
#define WS_CBH     0          // _Float16[1024*256]  = 524288 B
#define WS_CNORM   524288     // float[1024]
#define WS_COUNTS  528384     // float[1024]   <- zero-init region starts here
#define WS_SE      532480     // float[256]
#define WS_SZ      533504     // float[256]
#define WS_SCAL    534528     // float[4]: 0=Se2, 1=Sz2, 2=loss_acc
#define WS_IDX     534544     // int[32768]
#define ZERO_FLOATS 1540      // counts + s_e + s_z + scalars

#define CHUNK_GBYTES 16384u   // 32 codes * 256 halves * 2B in global
#define CHUNK_LDSB   16896u   // 32 rows * 528B (264-half padded rows) in LDS

typedef __attribute__((address_space(3))) char lds_char_t;
static __device__ __forceinline__ uint32_t lds_byte_off(const void* p) {
    return (uint32_t)(uintptr_t)(lds_char_t*)p;
}

// Issue one 32-code codebook chunk (16 KB) as 4x async b128 loads per thread,
// LDS-direct (no VGPR round-trip), tracked on ASYNCcnt.
static __device__ __forceinline__ void issue_chunk_async(const char* gsrc,
                                                         uint32_t ldsbase, int tid) {
    #pragma unroll
    for (int it = 0; it < 4; ++it) {
        int seg = it * 256 + tid;                 // 16B segment id in [0,1024)
        uint32_t loff = ldsbase + (uint32_t)((seg >> 5) * 528 + (seg & 31) * 16);
        uint32_t goff = (uint32_t)(seg * 16);
        asm volatile("global_load_async_to_lds_b128 %0, %1, %2"
                     :: "v"(loff), "v"(goff), "s"(gsrc) : "memory");
    }
}

// ---------------- init: zero atomic accumulators ----------------
__global__ __launch_bounds__(256) void vq_init(float* base) {
    int i = blockIdx.x * 256 + threadIdx.x;
    if (i < ZERO_FLOATS) base[i] = 0.0f;
}

// ---------------- codebook prep: f16 convert (scaled), row norms, Se2 ----------------
__global__ __launch_bounds__(256) void cb_prep(const float* __restrict__ cb,
                                               _Float16* __restrict__ cbh,
                                               float* __restrict__ cnorm,
                                               float* __restrict__ se2) {
    __shared__ float red[8];
    int k = blockIdx.x, t = threadIdx.x;
    float e = cb[k * EMB_D + t];
    cbh[k * EMB_D + t] = (_Float16)(e * CB_SCALE);
    float s = e * e;
    #pragma unroll
    for (int m = 16; m >= 1; m >>= 1) s += __shfl_xor(s, m, 32);
    if ((t & 31) == 0) red[t >> 5] = s;
    __syncthreads();
    if (t == 0) {
        float tot = 0.f;
        #pragma unroll
        for (int i = 0; i < 8; ++i) tot += red[i];
        cnorm[k] = tot;
        atomicAdd(se2, tot);
    }
}

// ---------------- codebook column sums: s_e[d] = sum_k e[k][d] ----------------
__global__ __launch_bounds__(256) void cb_colsum(const float* __restrict__ cb,
                                                 float* __restrict__ s_e) {
    __shared__ float red[8];
    int d = blockIdx.x, t = threadIdx.x;
    float s = 0.f;
    #pragma unroll
    for (int j = 0; j < 4; ++j) s += cb[(t + j * 256) * EMB_D + d];
    #pragma unroll
    for (int m = 16; m >= 1; m >>= 1) s += __shfl_xor(s, m, 32);
    if ((t & 31) == 0) red[t >> 5] = s;
    __syncthreads();
    if (t == 0) {
        float tot = 0.f;
        #pragma unroll
        for (int i = 0; i < 8; ++i) tot += red[i];
        s_e[d] = tot;
    }
}

// ---------------- z stats: per (b,c) plane sum and sumsq ----------------
__global__ __launch_bounds__(256) void z_stats(const float* __restrict__ z,
                                               float* __restrict__ s_z,
                                               float* __restrict__ sz2) {
    __shared__ float rs[8], rq[8];
    int p = blockIdx.x, t = threadIdx.x;         // p = b*256 + c
    const float4* zp = (const float4*)(z + (long)p * 1024);
    float4 v = zp[t];
    float s = v.x + v.y + v.z + v.w;
    float q = v.x * v.x + v.y * v.y + v.z * v.z + v.w * v.w;
    #pragma unroll
    for (int m = 16; m >= 1; m >>= 1) { s += __shfl_xor(s, m, 32); q += __shfl_xor(q, m, 32); }
    if ((t & 31) == 0) { rs[t >> 5] = s; rq[t >> 5] = q; }
    __syncthreads();
    if (t == 0) {
        float ts = 0.f, tq = 0.f;
        #pragma unroll
        for (int i = 0; i < 8; ++i) { ts += rs[i]; tq += rq[i]; }
        atomicAdd(&s_z[p & 255], ts);
        atomicAdd(sz2, tq);
    }
}

// ---------------- main WMMA distance + argmin kernel ----------------
// Grid: 512 blocks x 256 threads. Each WG: 64 rows x all 1024 codes.
// 8 waves = 4 M-tiles x 2 code-parity halves. D-loop = 8 x wmma_f32_16x16x32_f16.
// Codebook chunks are double-buffered in LDS and prefetched with
// GLOBAL_LOAD_ASYNC_TO_LDS_B128 one chunk ahead of compute (ASYNCcnt-tracked).
__global__ __launch_bounds__(256) void vq_argmin(const float* __restrict__ z,
                                                 const _Float16* __restrict__ cbh,
                                                 const float* __restrict__ cnorm,
                                                 float* __restrict__ counts,
                                                 int* __restrict__ idx_ws,
                                                 float* __restrict__ idx_out) {
    // zt (z staging) and the double-buffered codebook chunks are the same size
    // (16896 halves); zt is dead once A-fragments are in registers.
    __shared__ union {
        _Float16 zt[64][264];        // z tile, f16, padded rows
        _Float16 cbt[2][32][264];    // 2 x 32-code chunk, f16 scaled
    } u;
    __shared__ float cn_s[KCODES];
    __shared__ float rv[4][2][16];
    __shared__ int   ri[4][2][16];

    const int tid    = threadIdx.x;
    const int lane   = tid & 31;
    const int wave   = tid >> 5;
    const int mtile  = wave >> 1;
    const int parity = wave & 1;
    const int l15    = lane & 15;
    const int hh     = lane >> 4;       // which 16-lane half
    const int row_base = blockIdx.x * 64;
    const int b   = row_base >> 10;
    const int inb = row_base & 1023;
    const long zbase = (long)b * 262144 + inb;

    #pragma unroll
    for (int i = 0; i < 4; ++i) cn_s[tid + i * 256] = cnorm[tid + i * 256];

    // stage 64-row z tile into LDS as f16 (coalesced global reads along n)
    #pragma unroll 4
    for (int it = 0; it < 64; ++it) {
        int l = it * 256 + tid;
        int c = l >> 6;
        int ln = l & 63;
        u.zt[ln][c] = (_Float16)z[zbase + (long)c * 1024 + ln];
    }
    __syncthreads();

    // load this wave's A fragments once (16 rows x 256 d)
    v16h afrag[8];
    {
        const int arow = mtile * 16 + l15;
        #pragma unroll
        for (int kf = 0; kf < 8; ++kf) {
            int d0 = kf * 32 + hh * 8;
            v8h lo = *(const v8h*)&u.zt[arow][d0];
            v8h hi = *(const v8h*)&u.zt[arow][d0 + 16];
            v16h a;
            #pragma unroll
            for (int i = 0; i < 8; ++i) { a[i] = lo[i]; a[i + 8] = hi[i]; }
            afrag[kf] = a;
        }
    }
    __syncthreads();   // zt fully consumed by ALL waves -> safe to overwrite with chunks

    float bestv[8];
    int   besti[8];
    #pragma unroll
    for (int r = 0; r < 8; ++r) { bestv[r] = 3.4e38f; besti[r] = 0x7fffffff; }

    const char* cbbase = (const char*)cbh;
    const uint32_t cbt_off = lds_byte_off(&u.cbt[0][0][0]);

    // prefetch chunk 0 into buffer 0
    issue_chunk_async(cbbase, cbt_off, tid);

    for (int ch = 0; ch < 32; ++ch) {
        const int buf = ch & 1;
        if (ch + 1 < 32) {
            // prefetch next chunk into the other buffer, then wait only for
            // the current chunk (async loads complete in order per wave)
            issue_chunk_async(cbbase + (uint32_t)(ch + 1) * CHUNK_GBYTES,
                              cbt_off + (uint32_t)(buf ^ 1) * CHUNK_LDSB, tid);
            asm volatile("s_wait_asynccnt 0x4" ::: "memory");
        } else {
            asm volatile("s_wait_asynccnt 0x0" ::: "memory");
        }
        __syncthreads();   // current chunk visible to all waves

        const int clc = parity * 16 + l15;
        v8f acc = {0.f, 0.f, 0.f, 0.f, 0.f, 0.f, 0.f, 0.f};
        #pragma unroll
        for (int kf = 0; kf < 8; ++kf) {
            int d0 = kf * 32 + hh * 8;
            v8h lo = *(const v8h*)&u.cbt[buf][clc][d0];
            v8h hi = *(const v8h*)&u.cbt[buf][clc][d0 + 16];
            v16h bfr;
            #pragma unroll
            for (int i = 0; i < 8; ++i) { bfr[i] = lo[i]; bfr[i + 8] = hi[i]; }
            acc = __builtin_amdgcn_wmma_f32_16x16x32_f16(
                false, afrag[kf], false, bfr, (short)0, acc, false, false);
        }

        const int code = ch * 32 + parity * 16 + l15;
        const float cn = cn_s[code];
        #pragma unroll
        for (int r = 0; r < 8; ++r) {
            float v = cn - acc[r] * (2.0f / CB_SCALE);   // ||e||^2 - 2 z.e  (+||z||^2 const)
            if (v < bestv[r] || (v == bestv[r] && code < besti[r])) { bestv[r] = v; besti[r] = code; }
        }
        __syncthreads();   // all waves done reading buf before it is refilled
    }

    // reduce argmin across the 16 lanes holding each row
    #pragma unroll
    for (int r = 0; r < 8; ++r) {
        float v = bestv[r];
        int   i = besti[r];
        #pragma unroll
        for (int m = 8; m >= 1; m >>= 1) {
            float ov = __shfl_xor(v, m, 16);
            int   oi = __shfl_xor(i, m, 16);
            if (ov < v || (ov == v && oi < i)) { v = ov; i = oi; }
        }
        bestv[r] = v; besti[r] = i;
    }
    if (l15 == 0) {
        #pragma unroll
        for (int r = 0; r < 8; ++r) {
            rv[mtile][parity][hh * 8 + r] = bestv[r];
            ri[mtile][parity][hh * 8 + r] = besti[r];
        }
    }
    __syncthreads();

    // merge the two parity halves, emit idx + histogram
    if (tid < 64) {
        int mt = tid >> 4, row = tid & 15;
        float v0 = rv[mt][0][row]; int i0 = ri[mt][0][row];
        float v1 = rv[mt][1][row]; int i1 = ri[mt][1][row];
        int bi = (v1 < v0 || (v1 == v0 && i1 < i0)) ? i1 : i0;
        int n = row_base + mt * 16 + row;
        idx_ws[n]  = bi;
        idx_out[n] = (float)bi;
        atomicAdd(&counts[bi], 1.0f);
    }
}

// ---------------- gather z_q (fp32, NCHW), accumulate loss ----------------
__global__ __launch_bounds__(256) void vq_gather(const float* __restrict__ z,
                                                 const float* __restrict__ cb,
                                                 const int* __restrict__ idx_ws,
                                                 float* __restrict__ zq_out,
                                                 float* __restrict__ loss_acc) {
    __shared__ float red[8];
    int rg = blockIdx.x, tid = threadIdx.x;
    int w = tid & 31, g = tid >> 5;
    int b   = (rg * 32) >> 10;
    int inb = (rg * 32) & 1023;
    long base = (long)b * 262144 + inb + w;
    int n = rg * 32 + w;
    const float* crow = cb + (long)idx_ws[n] * EMB_D;
    float ls = 0.f;
    #pragma unroll 8
    for (int j = 0; j < 32; ++j) {
        int c = g * 32 + j;
        float e  = crow[c];
        long a   = base + (long)c * 1024;
        float zv = z[a];
        zq_out[a] = e;                 // straight-through forward value = z_q
        float d = e - zv;
        ls += d * d;
    }
    #pragma unroll
    for (int m = 16; m >= 1; m >>= 1) ls += __shfl_xor(ls, m, 32);
    if ((tid & 31) == 0) red[tid >> 5] = ls;
    __syncthreads();
    if (tid == 0) {
        float t = 0.f;
        #pragma unroll
        for (int i = 0; i < 8; ++i) t += red[i];
        atomicAdd(loss_acc, t);
    }
}

// ---------------- finalize scalars ----------------
__global__ __launch_bounds__(256) void vq_final(const float* __restrict__ counts,
                                                const float* __restrict__ s_e,
                                                const float* __restrict__ s_z,
                                                const float* __restrict__ scal,
                                                float* __restrict__ out) {
    __shared__ float rh[8], rd[8];
    int t = threadIdx.x;
    float h = 0.f;
    #pragma unroll
    for (int j = 0; j < 4; ++j) {
        float p = counts[t + j * 256] * (1.0f / 32768.0f);
        h += p * logf(p + 1e-10f);
    }
    float dp = s_e[t] * s_z[t];
    #pragma unroll
    for (int m = 16; m >= 1; m >>= 1) { h += __shfl_xor(h, m, 32); dp += __shfl_xor(dp, m, 32); }
    if ((t & 31) == 0) { rh[t >> 5] = h; rd[t >> 5] = dp; }
    __syncthreads();
    if (t == 0) {
        float H = 0.f, DOT = 0.f;
        #pragma unroll
        for (int i = 0; i < 8; ++i) { H += rh[i]; DOT += rd[i]; }
        float Se2 = scal[0], Sz2 = scal[1], L = scal[2];
        float md = (1024.0f * Sz2 + 32768.0f * Se2 - 2.0f * DOT) * (1.0f / (32768.0f * 1024.0f));
        out[8388608] = (1.0f + BETA_F) * L * (1.0f / 8388608.0f);  // loss
        out[8388609] = expf(-H);                                   // perplexity
        out[8421378] = md;                                         // mean_distance
    }
}

extern "C" void kernel_launch(void* const* d_in, const int* in_sizes, int n_in,
                              void* d_out, int out_size, void* d_ws, size_t ws_size,
                              hipStream_t stream) {
    (void)in_sizes; (void)n_in; (void)out_size; (void)ws_size;
    const float* z  = (const float*)d_in[0];
    const float* cb = (const float*)d_in[1];
    float* out = (float*)d_out;
    char* ws = (char*)d_ws;

    _Float16* cbh  = (_Float16*)(ws + WS_CBH);
    float* cnorm   = (float*)(ws + WS_CNORM);
    float* counts  = (float*)(ws + WS_COUNTS);
    float* s_e     = (float*)(ws + WS_SE);
    float* s_z     = (float*)(ws + WS_SZ);
    float* scal    = (float*)(ws + WS_SCAL);
    int*   idxw    = (int*)(ws + WS_IDX);

    float* zq   = out;               // [8388608] NCHW
    float* idxF = out + 8388610;     // [32768] idx as float

    vq_init  <<<7,    256, 0, stream>>>(counts);
    cb_prep  <<<1024, 256, 0, stream>>>(cb, cbh, cnorm, &scal[0]);
    cb_colsum<<<256,  256, 0, stream>>>(cb, s_e);
    z_stats  <<<8192, 256, 0, stream>>>(z, s_z, &scal[1]);
    vq_argmin<<<512,  256, 0, stream>>>(z, cbh, cnorm, counts, idxw, idxF);
    vq_gather<<<1024, 256, 0, stream>>>(z, cb, idxw, zq, &scal[2]);
    vq_final <<<1,    256, 0, stream>>>(counts, s_e, s_z, scal, out);
}
